// GroupGNN_Layer_81956565942353
// MI455X (gfx1250) — compile-verified, matrix operand order checked by reading, not run on hardware
//
#include <hip/hip_runtime.h>

#define NODE_H 128
#define EDGE_H 32
#define IN1    288   // 2*NODE_H + EDGE_H
#define KC1    9     // 288 / 32
#define ROWH1  296   // padded row stride (halves), 592B = 37*16
#define IN2    256   // NODE_H + GNN_H
#define KC2    8
#define ROWH2  264   // 528B = 33*16
#define WAVES  4     // waves per block (128 threads)

typedef __attribute__((ext_vector_type(16))) __bf16 v16bf;
typedef __attribute__((ext_vector_type(8)))  float  v8f;

union FragB16 {
    uint4 q[2];
    v16bf v;
};

__device__ __forceinline__ unsigned short f2bf(float f) {
    union { float f; unsigned u; } c; c.f = f;
    unsigned u = c.u;
    unsigned r = u + 0x7FFFu + ((u >> 16) & 1u);   // round-to-nearest-even
    return (unsigned short)(r >> 16);
}

// ---------------- prep: zero sums/counts, transpose+convert weights -------
__global__ void gnn_prep_kernel(const float* __restrict__ W1,
                                const float* __restrict__ W2,
                                unsigned short* __restrict__ w1t,
                                unsigned short* __restrict__ w2t,
                                float* __restrict__ sums,
                                float* __restrict__ cnt,
                                int n_nodes) {
    long long i = (long long)blockIdx.x * blockDim.x + threadIdx.x;
    long long total = (long long)n_nodes * NODE_H;
    if (i < total) sums[i] = 0.0f;
    if (i < n_nodes) cnt[i] = 0.0f;
    if (i < 128 * IN1) {
        int n = (int)(i / IN1), k = (int)(i % IN1);
        w1t[i] = f2bf(W1[(long long)k * 128 + n]);   // W1T[n][k] = W1[k][n]
    }
    if (i < 128 * IN2) {
        int n = (int)(i / IN2), k = (int)(i % IN2);
        w2t[i] = f2bf(W2[(long long)k * 128 + n]);
    }
}

// ------------- phase 1: fused edge MLP (WMMA) + scatter-add ---------------
__global__ __launch_bounds__(32 * WAVES)
void gnn_edge_kernel(const float* __restrict__ x,
                     const float* __restrict__ ea,
                     const float* __restrict__ b1,
                     const long long* __restrict__ eidx,
                     const unsigned short* __restrict__ w1t,
                     float* __restrict__ sums,
                     float* __restrict__ cnt,
                     int n_edges) {
    __shared__ __align__(16) unsigned short lds[WAVES * 16 * ROWH1];
    __shared__ int tcol[WAVES * 16];

    const int lane = threadIdx.x & 31;
    const int wv   = threadIdx.x >> 5;
    const int tile = blockIdx.x * WAVES + wv;
    const int ntiles = (n_edges + 15) >> 4;
    if (tile >= ntiles) return;

    const long long e0 = (long long)tile * 16;
    unsigned short* A = &lds[wv * 16 * ROWH1];
    const long long* rowI = eidx;
    const long long* colI = eidx + n_edges;

    // destination indices + degree counts (one lane per edge)
    if (lane < 16) {
        long long e = e0 + lane;
        int cI = (e < n_edges) ? (int)colI[e] : -1;
        tcol[wv * 16 + lane] = cI;
        if (cI >= 0) atomicAdd(&cnt[cI], 1.0f);
    }

    // stage A = [x[row] | x[col] | edge_attr] as bf16, one row per edge
    for (int r = 0; r < 16; ++r) {
        long long e = e0 + r;
        if (e < n_edges) {
            long long ri = rowI[e];
            long long ci = colI[e];
            #pragma unroll
            for (int i = 0; i < KC1; ++i) {            // 9*32 = 288 cols
                int c = lane + (i << 5);
                float val;
                if (c < NODE_H)          val = x[ri * NODE_H + c];
                else if (c < 2 * NODE_H) val = x[ci * NODE_H + (c - NODE_H)];
                else                     val = ea[e * EDGE_H + (c - 2 * NODE_H)];
                A[r * ROWH1 + c] = f2bf(val);
            }
        } else {
            #pragma unroll
            for (int i = 0; i < KC1; ++i) { int c = lane + (i << 5); A[r * ROWH1 + c] = 0; }
        }
    }

    const int nlo   = lane & 15;
    const int khalf = (lane >> 4) << 3;    // K sub-offset: 0 or 8

    // init accumulators with bias (C of first WMMA)
    v8f acc[8];
    #pragma unroll
    for (int nt = 0; nt < 8; ++nt) {
        float bv = b1[nt * 16 + nlo];
        v8f a;
        #pragma unroll
        for (int j = 0; j < 8; ++j) a[j] = bv;
        acc[nt] = a;
    }

    const unsigned short* Arow = A + nlo * ROWH1;      // A: row m = lane&15
    for (int kc = 0; kc < KC1; ++kc) {
        FragB16 af;
        int ko = kc * 32 + khalf;
        af.q[0] = *(const uint4*)(Arow + ko);
        af.q[1] = *(const uint4*)(Arow + ko + 16);
        #pragma unroll
        for (int nt = 0; nt < 8; ++nt) {
            FragB16 bf_;
            const unsigned short* Brow = w1t + (long long)(nt * 16 + nlo) * IN1;
            bf_.q[0] = *(const uint4*)(Brow + ko);
            bf_.q[1] = *(const uint4*)(Brow + ko + 16);
            acc[nt] = __builtin_amdgcn_wmma_f32_16x16x32_bf16(
                false, af.v, false, bf_.v, (short)0, acc[nt], false, false);
        }
    }

    // epilogue: ReLU + scatter-add into per-node sums (L2-resident atomics)
    const int mbase = (lane >> 4) << 3;
    #pragma unroll
    for (int nt = 0; nt < 8; ++nt) {
        int c = nt * 16 + nlo;
        #pragma unroll
        for (int j = 0; j < 8; ++j) {
            int m = mbase + j;
            int cI = tcol[wv * 16 + m];
            if (cI >= 0) {
                float v = fmaxf(acc[nt][j], 0.0f);
                atomicAdd(&sums[(long long)cI * NODE_H + c], v);
            }
        }
    }
}

// ------------- phase 2: mean + fused node MLP (WMMA), in-place in d_out ---
__global__ __launch_bounds__(32 * WAVES)
void gnn_node_kernel(const float* __restrict__ x,
                     const float* __restrict__ b2,
                     const unsigned short* __restrict__ w2t,
                     const float* __restrict__ cnt,
                     float* __restrict__ out,   // in: sums, out: result (disjoint rows per tile)
                     int n_nodes) {
    __shared__ __align__(16) unsigned short lds[WAVES * 16 * ROWH2];

    const int lane = threadIdx.x & 31;
    const int wv   = threadIdx.x >> 5;
    const int tile = blockIdx.x * WAVES + wv;
    const int ntiles = (n_nodes + 15) >> 4;
    if (tile >= ntiles) return;

    const int v0 = tile * 16;
    unsigned short* A = &lds[wv * 16 * ROWH2];

    // stage A = [x[v] | agg[v]] as bf16
    for (int r = 0; r < 16; ++r) {
        int v = v0 + r;
        if (v < n_nodes) {
            float inv = 1.0f / fmaxf(cnt[v], 1.0f);
            #pragma unroll
            for (int i = 0; i < KC2; ++i) {            // 8*32 = 256 cols
                int c = lane + (i << 5);
                float val = (c < NODE_H)
                          ? x[(long long)v * NODE_H + c]
                          : out[(long long)v * NODE_H + (c - NODE_H)] * inv;
                A[r * ROWH2 + c] = f2bf(val);
            }
        } else {
            #pragma unroll
            for (int i = 0; i < KC2; ++i) { int c = lane + (i << 5); A[r * ROWH2 + c] = 0; }
        }
    }

    const int nlo   = lane & 15;
    const int khalf = (lane >> 4) << 3;

    v8f acc[8];
    #pragma unroll
    for (int nt = 0; nt < 8; ++nt) {
        float bv = b2[nt * 16 + nlo];
        v8f a;
        #pragma unroll
        for (int j = 0; j < 8; ++j) a[j] = bv;
        acc[nt] = a;
    }

    const unsigned short* Arow = A + nlo * ROWH2;
    for (int kc = 0; kc < KC2; ++kc) {
        FragB16 af;
        int ko = kc * 32 + khalf;
        af.q[0] = *(const uint4*)(Arow + ko);
        af.q[1] = *(const uint4*)(Arow + ko + 16);
        #pragma unroll
        for (int nt = 0; nt < 8; ++nt) {
            FragB16 bf_;
            const unsigned short* Brow = w2t + (long long)(nt * 16 + nlo) * IN2;
            bf_.q[0] = *(const uint4*)(Brow + ko);
            bf_.q[1] = *(const uint4*)(Brow + ko + 16);
            acc[nt] = __builtin_amdgcn_wmma_f32_16x16x32_bf16(
                false, af.v, false, bf_.v, (short)0, acc[nt], false, false);
        }
    }

    const int mbase = (lane >> 4) << 3;
    #pragma unroll
    for (int nt = 0; nt < 8; ++nt) {
        int c = nt * 16 + nlo;
        #pragma unroll
        for (int j = 0; j < 8; ++j) {
            int v = v0 + mbase + j;
            if (v < n_nodes) out[(long long)v * NODE_H + c] = fmaxf(acc[nt][j], 0.0f);
        }
    }
}

extern "C" void kernel_launch(void* const* d_in, const int* in_sizes, int n_in,
                              void* d_out, int out_size, void* d_ws, size_t ws_size,
                              hipStream_t stream) {
    const float*     x    = (const float*)d_in[0];
    const float*     ea   = (const float*)d_in[1];
    const float*     W1   = (const float*)d_in[2];
    const float*     b1   = (const float*)d_in[3];
    const float*     W2   = (const float*)d_in[4];
    const float*     b2   = (const float*)d_in[5];
    const long long* eidx = (const long long*)d_in[6];
    float* out = (float*)d_out;

    const int NN = in_sizes[0] / NODE_H;   // 50000
    const int NE = in_sizes[6] / 2;        // 800000

    // workspace: bf16 W1^T | bf16 W2^T | f32 counts   (~340 KB)
    unsigned short* w1t = (unsigned short*)d_ws;
    unsigned short* w2t = w1t + 128 * IN1;
    float*          cnt = (float*)(w2t + 128 * IN2);

    long long prepTot = (long long)NN * NODE_H;
    int pb = (int)((prepTot + 255) / 256);
    gnn_prep_kernel<<<pb, 256, 0, stream>>>(W1, W2, w1t, w2t, out, cnt, NN);

    int etiles = (NE + 15) / 16;
    gnn_edge_kernel<<<(etiles + WAVES - 1) / WAVES, 32 * WAVES, 0, stream>>>(
        x, ea, b1, eidx, w1t, out, cnt, NE);

    int vtiles = (NN + 15) / 16;
    gnn_node_kernel<<<(vtiles + WAVES - 1) / WAVES, 32 * WAVES, 0, stream>>>(
        x, b2, w2t, cnt, out, NN);
}
